// NanoGPT_62191126446398
// MI455X (gfx1250) — matrix operand order repaired and checked by reading, not compiled, not run on hardware
//
#include <hip/hip_runtime.h>
#include <cstdint>
#include <cstddef>

#define LYR 6
#define DM  1024
#define NH  16
#define SQ  1024
#define VOC 32000
#define BSZ 2
#define HDM 64
#define MT  (BSZ*SQ)   /* 2048 token rows */

#define USE_ASYNC_ASM 1

typedef __bf16 bf16;
typedef __bf16 v16bf __attribute__((ext_vector_type(16)));
typedef float  v8f   __attribute__((ext_vector_type(8)));
typedef unsigned int v8u __attribute__((ext_vector_type(8)));
typedef unsigned int v4u __attribute__((ext_vector_type(4)));
typedef int v8i __attribute__((ext_vector_type(8)));
typedef int v4i __attribute__((ext_vector_type(4)));

static __device__ __forceinline__ bf16 f2bf(float f) {
  unsigned u = __builtin_bit_cast(unsigned, f);
  u += 0x7FFFu + ((u >> 16) & 1u);                 // round-to-nearest-even
  unsigned short h = (unsigned short)(u >> 16);
  return __builtin_bit_cast(bf16, h);
}
static __device__ __forceinline__ v8f vzero8() {
  v8f z;
  #pragma unroll
  for (int i = 0; i < 8; ++i) z[i] = 0.0f;
  return z;
}
static __device__ __forceinline__ float gelu_exact(float x) {
  return 0.5f * x * (1.0f + erff(x * 0.70710678118654752f));
}

// ---- 16-lane butterfly reductions (VALU v_permlane16, no LDS round trips) ----
#if __has_builtin(__builtin_amdgcn_permlane16)
static __device__ __forceinline__ float permx16(float v, unsigned lo, unsigned hi) {
  unsigned u = __builtin_bit_cast(unsigned, v);
  unsigned r = __builtin_amdgcn_permlane16(u, u, lo, hi, false, false);
  return __builtin_bit_cast(float, r);
}
static __device__ __forceinline__ float rowmax16(float v) {
  v = fmaxf(v, permx16(v, 0x67452301u, 0xEFCDAB89u));   // xor 1
  v = fmaxf(v, permx16(v, 0x54761032u, 0xDCFE98BAu));   // xor 2
  v = fmaxf(v, permx16(v, 0x32107654u, 0xBA98FEDCu));   // xor 4
  v = fmaxf(v, permx16(v, 0xFEDCBA98u, 0x76543210u));   // xor 8
  return v;
}
static __device__ __forceinline__ float rowsum16(float v) {
  v += permx16(v, 0x67452301u, 0xEFCDAB89u);
  v += permx16(v, 0x54761032u, 0xDCFE98BAu);
  v += permx16(v, 0x32107654u, 0xBA98FEDCu);
  v += permx16(v, 0xFEDCBA98u, 0x76543210u);
  return v;
}
#else
static __device__ __forceinline__ float rowmax16(float v) {
  #pragma unroll
  for (int d = 1; d < 16; d <<= 1) v = fmaxf(v, __shfl_xor(v, d, 32));
  return v;
}
static __device__ __forceinline__ float rowsum16(float v) {
  #pragma unroll
  for (int d = 1; d < 16; d <<= 1) v += __shfl_xor(v, d, 32);
  return v;
}
#endif

// ---- Tensor Data Mover: 2D tile (bf16) global -> LDS with row padding ----
#if __has_builtin(__builtin_amdgcn_tensor_load_to_lds)
#define HAVE_TDM 1
// tile_d0 = contiguous elements per row, tile_d1 = rows,
// row_stride = elements between rows, pad codes per ISA D# group1.
static __device__ __forceinline__ void tdm_load_2d(
    unsigned lds_addr, unsigned long long gaddr,
    unsigned tile_d0, unsigned tile_d1, unsigned long long row_stride,
    unsigned pad_interval_code, unsigned pad_amount_code) {
  v4u g0;
  g0[0] = 1u;                                            // count=1, user mode
  g0[1] = lds_addr;                                      // LDS byte address
  g0[2] = (unsigned)(gaddr & 0xFFFFFFFFull);             // global_addr[31:0]
  g0[3] = (unsigned)((gaddr >> 32) & 0x01FFFFFFull)      // global_addr[56:32]
          | (2u << 30);                                  // type = 2 ("image")
  unsigned w0 = (1u << 16);                              // data_size = 2 bytes
  w0 |= (1u << 20) | (pad_interval_code << 22) | (pad_amount_code << 25);
  v8i g1;
  g1[0] = (int)w0;
  g1[1] = (int)((tile_d0 & 0xFFFFu) << 16);              // tensor_dim0[15:0]
  g1[2] = (int)(((tile_d0 >> 16) & 0xFFFFu) |            // tensor_dim0[31:16]
                ((tile_d1 & 0xFFFFu) << 16));            // tensor_dim1[15:0]
  g1[3] = (int)(((tile_d1 >> 16) & 0xFFFFu) |            // tensor_dim1[31:16]
                ((tile_d0 & 0xFFFFu) << 16));            // tile_dim0
  g1[4] = (int)(tile_d1 & 0xFFFFu);                      // tile_dim1 (dim2=0)
  g1[5] = (int)(row_stride & 0xFFFFFFFFull);             // dim0_stride[31:0]
  g1[6] = (int)((row_stride >> 32) & 0xFFFFull);         // dim0_stride[47:32]
  g1[7] = 0;
  v4i z4;  z4[0] = 0; z4[1] = 0; z4[2] = 0; z4[3] = 0;
  v8i z8;
  #pragma unroll
  for (int i = 0; i < 8; ++i) z8[i] = 0;
  // amdgpu-toolchain (clang-23) 6-arg form: groups 0..3 + extension + cpol
  __builtin_amdgcn_tensor_load_to_lds(g0, g1, z4, z4, z8, 0);
}
#endif

// ---------------------------------------------------------------- convert
__global__ __launch_bounds__(256) void f32_to_bf16_kernel(
    const float* __restrict__ in, bf16* __restrict__ out, long n) {
  long i = (long)blockIdx.x * blockDim.x + threadIdx.x;
  long stride = (long)gridDim.x * blockDim.x;
  for (; i < n; i += stride) out[i] = f2bf(in[i]);
}

// ---------------------------------------------------------------- embedding
__global__ __launch_bounds__(256) void embed_kernel(
    const int* __restrict__ X, const float* __restrict__ TE,
    const float* __restrict__ PE, float* __restrict__ H) {
  int row = blockIdx.x, t = threadIdx.x;
  int s = row % SQ;
  int tok = X[row];
  const float* te = TE + (size_t)tok * DM;
  const float* pe = PE + (size_t)s * DM;
  float* h = H + (size_t)row * DM;
  #pragma unroll
  for (int i = 0; i < 4; ++i) { int d = t + i * 256; h[d] = te[d] + pe[d]; }
}

// ---------------------------------------------------------------- layernorm -> bf16
__global__ __launch_bounds__(256) void ln_bf16_kernel(
    const float* __restrict__ X, const float* __restrict__ Sc,
    const float* __restrict__ Bi, bf16* __restrict__ O) {
  __shared__ float sred[256], sred2[256];
  int row = blockIdx.x, t = threadIdx.x;
  const float* x = X + (size_t)row * DM;
  float s = 0.0f, s2 = 0.0f;
  #pragma unroll
  for (int i = 0; i < 4; ++i) { float v = x[t + i * 256]; s += v; s2 += v * v; }
  sred[t] = s; sred2[t] = s2;
  __syncthreads();
  for (int d = 128; d > 0; d >>= 1) {
    if (t < d) { sred[t] += sred[t + d]; sred2[t] += sred2[t + d]; }
    __syncthreads();
  }
  float mu  = sred[0] * (1.0f / DM);
  float var = sred2[0] * (1.0f / DM) - mu * mu;
  float inv = rsqrtf(var + 1e-5f);
  bf16* o = O + (size_t)row * DM;
  #pragma unroll
  for (int i = 0; i < 4; ++i) {
    int d = t + i * 256;
    o[d] = f2bf((x[d] - mu) * inv * Sc[d] + Bi[d]);
  }
}

// ---------------------------------------------------------------- WMMA GEMM
// C[M,N] = A[M,K](bf16,rm) x B[K,N](bf16,rm) + bias
// MODE 0: out bf16          MODE 1: gelu, out bf16
// MODE 2: +Res, out f32     MODE 3: out f32
template <int MODE>
__global__ __launch_bounds__(256) void gemm_bf16_kernel(
    const bf16* __restrict__ A, const bf16* __restrict__ Bw,
    const float* __restrict__ bias, const float* __restrict__ Res,
    void* __restrict__ Out, int M, int N, int K) {
  __shared__ bf16 As[128 * 40];   // [m][k] pad 32->40
  __shared__ bf16 Bt[128 * 40];   // [n][k] pad 32->40 (transposed stage)
  const int t = threadIdx.x;
  const int lane = t & 31;
  const int wv = t >> 5;
  const int wRow = wv >> 2, wCol = wv & 3;
  const int mBase = blockIdx.y * 128, nBase = blockIdx.x * 128;
  const int lh = lane >> 4;        // lane half
  const int lm = lane & 15;

  v8f acc[4][2];
  #pragma unroll
  for (int i = 0; i < 4; ++i)
    #pragma unroll
    for (int j = 0; j < 2; ++j) acc[i][j] = vzero8();

  for (int kc = 0; kc < K; kc += 32) {
    __syncthreads();
#if defined(HAVE_TDM)
    if (wv == 0) {     // TDM: one DMA descriptor stages the whole 128x32 A tile
      tdm_load_2d((unsigned)(uintptr_t)As,
                  (unsigned long long)(uintptr_t)(A + (size_t)mBase * K + kc),
                  32u, 128u, (unsigned long long)K,
                  /*pad every 16 DW*/ 3u, /*by 4 DW*/ 3u);
      __builtin_amdgcn_s_wait_tensorcnt(0);
    }
#else
    { // stage A 128x32 (coalesced b128)
      int r = t >> 2, c = (t & 3) * 8;
      *(uint4*)(&As[r * 40 + c]) =
          *(const uint4*)(A + (size_t)(mBase + r) * K + kc + c);
      *(uint4*)(&As[(r + 64) * 40 + c]) =
          *(const uint4*)(A + (size_t)(mBase + r + 64) * K + kc + c);
    }
#endif
    { // stage B 32x128 transposed -> Bt[n][k]
      int kr = t >> 4, nc = (t & 15) * 8;
      #pragma unroll
      for (int p = 0; p < 2; ++p) {
        int kk = kr + p * 16;
        uint4 v = *(const uint4*)(Bw + (size_t)(kc + kk) * N + nBase + nc);
        const bf16* e = (const bf16*)&v;
        #pragma unroll
        for (int i = 0; i < 8; ++i) Bt[(nc + i) * 40 + kk] = e[i];
      }
      if (kc + 32 < K)   // prefetch next B tile (global_prefetch_b8)
        __builtin_prefetch(Bw + (size_t)(kc + 32 + kr) * N + nBase + nc, 0, 1);
    }
    __syncthreads();

    v16bf bfrag[2];
    #pragma unroll
    for (int st = 0; st < 2; ++st) {           // B fragments (ISA B layout)
      v8u u;
      int n = wCol * 32 + st * 16 + lm;
      #pragma unroll
      for (int j = 0; j < 8; ++j) {
        int kk = lh * 8 + (j & 3) * 2 + (j >> 2) * 16;
        u[j] = *(const unsigned*)&Bt[n * 40 + kk];
      }
      bfrag[st] = __builtin_bit_cast(v16bf, u);
    }
    #pragma unroll
    for (int mt = 0; mt < 4; ++mt) {           // A fragments (ISA A layout)
      v8u u;
      int mr = wRow * 64 + mt * 16 + lm;
      #pragma unroll
      for (int j = 0; j < 8; ++j) {
        int kk = lh * 8 + (j & 3) * 2 + (j >> 2) * 16;
        u[j] = *(const unsigned*)&As[mr * 40 + kk];
      }
      v16bf afrag = __builtin_bit_cast(v16bf, u);
      acc[mt][0] = __builtin_amdgcn_wmma_f32_16x16x32_bf16(
          false, afrag, false, bfrag[0], (short)0, acc[mt][0], false, false);
      acc[mt][1] = __builtin_amdgcn_wmma_f32_16x16x32_bf16(
          false, afrag, false, bfrag[1], (short)0, acc[mt][1], false, false);
    }
  }

  #pragma unroll
  for (int mt = 0; mt < 4; ++mt)
    #pragma unroll
    for (int st = 0; st < 2; ++st)
      #pragma unroll
      for (int r = 0; r < 8; ++r) {
        int Mi = mBase + wRow * 64 + mt * 16 + r + 8 * lh;
        int Ni = nBase + wCol * 32 + st * 16 + lm;
        float v = acc[mt][st][r] + bias[Ni];
        if (MODE == 1) v = gelu_exact(v);
        if (MODE == 2) {
          v += Res[(size_t)Mi * N + Ni];
          ((float*)Out)[(size_t)Mi * N + Ni] = v;
        } else if (MODE == 3) {
          ((float*)Out)[(size_t)Mi * N + Ni] = v;
        } else {
          ((bf16*)Out)[(size_t)Mi * N + Ni] = f2bf(v);
        }
      }
}

// ---------------------------------------------------------------- flash attention (WMMA)
// qkv: [B, S, 3D] bf16 (q|k|v).  out: [B, S, D] bf16.
__global__ __launch_bounds__(256) void attn_kernel(
    const bf16* __restrict__ QKV, bf16* __restrict__ O) {
  __shared__ bf16 Qs[128 * 72];       // [q][d]
  __shared__ bf16 Ks[32 * 72];        // [key][d]  (== B-frag order for Q·Kt)
  __shared__ bf16 Vt[64 * 40];        // [d][key]  (== B-frag order for P·V)
  __shared__ bf16 Ps[8 * 16 * 40];    // per-wave P tile 16x32
  const int t = threadIdx.x, lane = t & 31, wv = t >> 5;
  const int lh = lane >> 4, lm = lane & 15;
  const int qb = blockIdx.x * 128;
  const int hh = blockIdx.y;
  const int bb = blockIdx.z;
  const size_t rs = 3 * DM;
  const size_t base = (size_t)bb * SQ * rs + (size_t)hh * HDM;

#if defined(HAVE_TDM)
  if (wv == 0) {          // TDM: stage Q block 128x64 (pad 32->36 DWORD rows)
    tdm_load_2d((unsigned)(uintptr_t)Qs,
                (unsigned long long)(uintptr_t)(QKV + base + (size_t)qb * rs),
                64u, 128u, (unsigned long long)rs,
                /*pad every 32 DW*/ 4u, /*by 4 DW*/ 3u);
    __builtin_amdgcn_s_wait_tensorcnt(0);
  }
#else
  { // stage Q block 128x64
    int r = t >> 3, c = (t & 7) * 8;
    #pragma unroll
    for (int p = 0; p < 4; ++p) {
      int rr = r + p * 32;
      *(uint4*)(&Qs[rr * 72 + c]) =
          *(const uint4*)(QKV + base + (size_t)(qb + rr) * rs + c);
    }
  }
#endif
  __syncthreads();

  v16bf qf[2];
  #pragma unroll
  for (int cf = 0; cf < 2; ++cf) {
    v8u u;
    #pragma unroll
    for (int j = 0; j < 8; ++j) {
      int kk = cf * 32 + lh * 8 + (j & 3) * 2 + (j >> 2) * 16;
      u[j] = *(const unsigned*)&Qs[(wv * 16 + lm) * 72 + kk];
    }
    qf[cf] = __builtin_bit_cast(v16bf, u);
  }

  float mrow[8], lrow[8];
  #pragma unroll
  for (int r = 0; r < 8; ++r) { mrow[r] = -3.0e38f; lrow[r] = 0.0f; }
  v8f oa[4];
  #pragma unroll
  for (int n = 0; n < 4; ++n) oa[n] = vzero8();

  const int kend = qb + 128;
  for (int kb = 0; kb < kend; kb += 32) {
    __syncthreads();
    {
      int kr = t >> 3, c = (t & 7) * 8;
#if USE_ASYNC_ASM
      { // async global->LDS: K block 32x64, one b128 per lane, ASYNCcnt-tracked
        unsigned loff = (unsigned)(uintptr_t)&Ks[kr * 72 + c];
        const bf16* gk = QKV + base + DM + (size_t)(kb + kr) * rs + c;
        asm volatile("global_load_async_to_lds_b128 %0, %1, off"
                     :: "v"(loff), "v"(gk) : "memory");
      }
#else
      *(uint4*)(&Ks[kr * 72 + c]) =
          *(const uint4*)(QKV + base + DM + (size_t)(kb + kr) * rs + c);
#endif
      // V block staged transposed (TDM/async cannot transpose)
      uint4 vv = *(const uint4*)(QKV + base + 2 * DM + (size_t)(kb + kr) * rs + c);
      const bf16* e = (const bf16*)&vv;
      #pragma unroll
      for (int i = 0; i < 8; ++i) Vt[(c + i) * 40 + kr] = e[i];
    }
#if USE_ASYNC_ASM
    asm volatile("s_wait_asynccnt 0x0" ::: "memory");
#endif
    __syncthreads();

    // scores: two 16x16 key sub-tiles, K-dim = 64 (2 WMMAs each)
    v8f s0 = vzero8(), s1 = vzero8();
    #pragma unroll
    for (int st = 0; st < 2; ++st) {
      v8f a = vzero8();
      #pragma unroll
      for (int cf = 0; cf < 2; ++cf) {
        v8u u;
        #pragma unroll
        for (int j = 0; j < 8; ++j) {
          int kk = lh * 8 + (j & 3) * 2 + (j >> 2) * 16;
          u[j] = *(const unsigned*)&Ks[(st * 16 + lm) * 72 + cf * 32 + kk];
        }
        v16bf kf = __builtin_bit_cast(v16bf, u);
        a = __builtin_amdgcn_wmma_f32_16x16x32_bf16(
            false, qf[cf], false, kf, (short)0, a, false, false);
      }
      if (st == 0) s0 = a; else s1 = a;
    }

    // online softmax (row = r + 8*lh, cols live on the 16-lane half)
    #pragma unroll
    for (int r = 0; r < 8; ++r) {
      int qi = qb + wv * 16 + r + 8 * lh;
      int k0 = kb + lm, k1 = kb + 16 + lm;
      float v0 = (k0 <= qi) ? s0[r] * 0.125f : -3.0e38f;
      float v1 = (k1 <= qi) ? s1[r] * 0.125f : -3.0e38f;
      float mx = rowmax16(fmaxf(v0, v1));
      float mnew  = fmaxf(mrow[r], mx);
      float alpha = __expf(mrow[r] - mnew);
      float p0 = __expf(v0 - mnew), p1 = __expf(v1 - mnew);
      float rsum = rowsum16(p0 + p1);
      lrow[r] = lrow[r] * alpha + rsum;
      mrow[r] = mnew;
      #pragma unroll
      for (int n = 0; n < 4; ++n) oa[n][r] *= alpha;
      int m = r + 8 * lh;
      Ps[(wv * 16 + m) * 40 + lm]      = f2bf(p0);
      Ps[(wv * 16 + m) * 40 + 16 + lm] = f2bf(p1);
    }

    // P (16x32) as A fragment, accumulate P·V over 4 output column tiles
    v8u pu;
    #pragma unroll
    for (int j = 0; j < 8; ++j) {
      int kk = lh * 8 + (j & 3) * 2 + (j >> 2) * 16;
      pu[j] = *(const unsigned*)&Ps[(wv * 16 + lm) * 40 + kk];
    }
    v16bf pf = __builtin_bit_cast(v16bf, pu);
    #pragma unroll
    for (int n = 0; n < 4; ++n) {
      v8u u;
      #pragma unroll
      for (int j = 0; j < 8; ++j) {
        int kk = lh * 8 + (j & 3) * 2 + (j >> 2) * 16;
        u[j] = *(const unsigned*)&Vt[(n * 16 + lm) * 40 + kk];
      }
      v16bf vf = __builtin_bit_cast(v16bf, u);
      oa[n] = __builtin_amdgcn_wmma_f32_16x16x32_bf16(
          false, pf, false, vf, (short)0, oa[n], false, false);
    }
  }

  #pragma unroll
  for (int n = 0; n < 4; ++n)
    #pragma unroll
    for (int r = 0; r < 8; ++r) {
      int m = r + 8 * lh;
      int row = qb + wv * 16 + m;
      int col = hh * HDM + n * 16 + lm;
      O[((size_t)bb * SQ + row) * DM + col] = f2bf(oa[n][r] / lrow[r]);
    }
}

// ---------------------------------------------------------------- launcher
extern "C" void kernel_launch(void* const* d_in, const int* in_sizes, int n_in,
                              void* d_out, int out_size, void* d_ws, size_t ws_size,
                              hipStream_t stream) {
  (void)in_sizes; (void)n_in; (void)out_size; (void)ws_size;
  const int*   x       = (const int*)  d_in[0];
  const float* tok_emb = (const float*)d_in[1];
  const float* pos_emb = (const float*)d_in[2];
  const float* qkv_w   = (const float*)d_in[3];
  const float* qkv_b   = (const float*)d_in[4];
  const float* proj_w  = (const float*)d_in[5];
  const float* proj_b  = (const float*)d_in[6];
  const float* ln1_s   = (const float*)d_in[7];
  const float* ln1_b   = (const float*)d_in[8];
  const float* ln2_s   = (const float*)d_in[9];
  const float* ln2_b   = (const float*)d_in[10];
  const float* mlp_w1  = (const float*)d_in[11];
  const float* mlp_b1  = (const float*)d_in[12];
  const float* mlp_w2  = (const float*)d_in[13];
  const float* mlp_b2  = (const float*)d_in[14];
  const float* lnf_s   = (const float*)d_in[15];
  const float* lnf_b   = (const float*)d_in[16];
  const float* head_w  = (const float*)d_in[17];
  const float* head_b  = (const float*)d_in[18];
  float* out = (float*)d_out;

  char* ws = (char*)d_ws;
  size_t off = 0;
  auto take = [&](size_t bytes) -> char* {
    char* p = ws + off;
    off += (bytes + 255) & ~(size_t)255;
    return p;
  };
  bf16*  wqkv  = (bf16*)take((size_t)LYR * DM * 3 * DM * 2);
  bf16*  wproj = (bf16*)take((size_t)LYR * DM * DM * 2);
  bf16*  wm1   = (bf16*)take((size_t)LYR * DM * 4 * DM * 2);
  bf16*  wm2   = (bf16*)take((size_t)LYR * 4 * DM * DM * 2);
  bf16*  whead = (bf16*)take((size_t)DM * VOC * 2);
  float* h     = (float*)take((size_t)MT * DM * 4);
  bf16*  xn    = (bf16*)take((size_t)MT * DM * 2);
  bf16*  qkv   = (bf16*)take((size_t)MT * 3 * DM * 2);
  bf16*  att   = (bf16*)take((size_t)MT * DM * 2);
  bf16*  mb    = (bf16*)take((size_t)MT * 4 * DM * 2);

  // weight precision cast (idempotent; rerun every launch)
  f32_to_bf16_kernel<<<2048, 256, 0, stream>>>(qkv_w,  wqkv,  (long)LYR * DM * 3 * DM);
  f32_to_bf16_kernel<<<2048, 256, 0, stream>>>(proj_w, wproj, (long)LYR * DM * DM);
  f32_to_bf16_kernel<<<2048, 256, 0, stream>>>(mlp_w1, wm1,   (long)LYR * DM * 4 * DM);
  f32_to_bf16_kernel<<<2048, 256, 0, stream>>>(mlp_w2, wm2,   (long)LYR * 4 * DM * DM);
  f32_to_bf16_kernel<<<2048, 256, 0, stream>>>(head_w, whead, (long)DM * VOC);

  embed_kernel<<<MT, 256, 0, stream>>>(x, tok_emb, pos_emb, h);

  for (int l = 0; l < LYR; ++l) {
    ln_bf16_kernel<<<MT, 256, 0, stream>>>(h, ln1_s + l * DM, ln1_b + l * DM, xn);
    gemm_bf16_kernel<0><<<dim3(3 * DM / 128, MT / 128), 256, 0, stream>>>(
        xn, wqkv + (size_t)l * DM * 3 * DM, qkv_b + (size_t)l * 3 * DM,
        nullptr, qkv, MT, 3 * DM, DM);
    attn_kernel<<<dim3(SQ / 128, NH, BSZ), 256, 0, stream>>>(qkv, att);
    gemm_bf16_kernel<2><<<dim3(DM / 128, MT / 128), 256, 0, stream>>>(
        att, wproj + (size_t)l * DM * DM, proj_b + (size_t)l * DM,
        h, h, MT, DM, DM);
    ln_bf16_kernel<<<MT, 256, 0, stream>>>(h, ln2_s + l * DM, ln2_b + l * DM, xn);
    gemm_bf16_kernel<1><<<dim3(4 * DM / 128, MT / 128), 256, 0, stream>>>(
        xn, wm1 + (size_t)l * DM * 4 * DM, mlp_b1 + (size_t)l * 4 * DM,
        nullptr, mb, MT, 4 * DM, DM);
    gemm_bf16_kernel<2><<<dim3(DM / 128, MT / 128), 256, 0, stream>>>(
        mb, wm2 + (size_t)l * 4 * DM * DM, mlp_b2 + (size_t)l * DM,
        h, h, MT, DM, 4 * DM);
  }

  ln_bf16_kernel<<<MT, 256, 0, stream>>>(h, lnf_s, lnf_b, xn);
  gemm_bf16_kernel<3><<<dim3(VOC / 128, MT / 128), 256, 0, stream>>>(
      xn, whead, head_b, nullptr, out, MT, VOC, DM);
}
